// Decoder_84026740179520
// MI455X (gfx1250) — matrix operand order repaired and checked by reading, not compiled, not run on hardware
//
#include <hip/hip_runtime.h>
#include <hip/hip_bf16.h>

// CDNA5 / gfx1250 WMMA types
typedef __attribute__((ext_vector_type(16))) __bf16 bf16x16;
typedef __attribute__((ext_vector_type(8)))  float  floatx8;

#define B_     32
#define TENC   128
#define TOUT   256
#define IDIM   512
#define ODIM   80
#define HID    1024
#define PRE    256
#define ATT    128
#define CCH    32
#define KATT   31
#define ZONE   0.1f
#define SCAL   2.0f

__device__ __forceinline__ float sigmf(float x){ return 1.0f/(1.0f+__expf(-x)); }

__device__ __forceinline__ floatx8 wmma_bf16(bf16x16 a, bf16x16 b, floatx8 c){
#if defined(__AMDGCN__)
  // v_wmma_f32_16x16x32_bf16: D = A(16x32) * B(32x16) + C, fp32 accumulate
  return __builtin_amdgcn_wmma_f32_16x16x32_bf16(false, a, false, b, (short)0, c, false, false);
#else
  (void)a; (void)b; return c;
#endif
}

// ---------------- one-time init kernels ----------------

__global__ void k_init(float* z0, float* c0, float* z1, float* c1, float* aw){
  int g = blockIdx.x*blockDim.x + threadIdx.x;
  if (g < B_*HID){ z0[g]=0.f; c0[g]=0.f; z1[g]=0.f; c1[g]=0.f; }
  if (g < B_*TENC) aw[g] = 1.0f/(float)TENC;
}

__global__ void k_biascomb(const float* a0,const float* b0,const float* a1,const float* b1,
                           float* bc0,float* bc1){
  int g = blockIdx.x*blockDim.x + threadIdx.x;
  if (g < 4*HID){ bc0[g]=a0[g]+b0[g]; bc1[g]=a1[g]+b1[g]; }
}

// enc_proj[b,t,a] = hs[b,t,:] @ Wenc + benc   (grid = B*TENC blocks, block = ATT)
__global__ void k_encproj(const float* __restrict__ hs, const float* __restrict__ We,
                          const float* __restrict__ be, float* __restrict__ enc){
  int bt = blockIdx.x; int a = threadIdx.x;
  float s = be[a];
  const float* h = hs + (size_t)bt*IDIM;
  for (int i = 0; i < IDIM; ++i) s += h[i]*We[i*ATT + a];
  enc[(size_t)bt*ATT + a] = s;
}

// Pack combined LSTM weight [W0 | W1] (N x Ktot) transposed into per-lane WMMA
// B-fragment order: Wpk[(ntile*ktiles + kt)*32 + lane][16] contiguous bf16.
// k order chosen here (A packs match): lane l -> column n = l&15; K = (l>>4)*16 + e.
__global__ void k_packW_lstm(__bf16* dst, const float* W0, int k0, const float* W1, int k1,
                             int ntiles, int ktiles){
  int gid = blockIdx.x*blockDim.x + threadIdx.x;
  int total = ntiles*ktiles*32;
  if (gid >= total) return;
  int lane = gid & 31;
  int tk = (gid >> 5) % ktiles;
  int tn = (gid >> 5) / ktiles;
  int n  = tn*16 + (lane & 15);
  int kb = tk*32 + (lane >> 4)*16;
  __bf16* d = dst + (size_t)gid*16;
  for (int e = 0; e < 16; ++e){
    int k = kb + e;
    float v = (k < k0) ? W0[(size_t)n*k0 + k] : W1[(size_t)n*k1 + (k - k0)];
    d[e] = (__bf16)v;
  }
}

// Pack conv1d weight (OIH, kernel 5) as GEMM B with K = Cin*5 (zero-padded to ktiles*32)
__global__ void k_packW_conv(__bf16* dst, const float* w, int Cin, int ntiles, int ktiles){
  int gid = blockIdx.x*blockDim.x + threadIdx.x;
  int total = ntiles*ktiles*32;
  if (gid >= total) return;
  int lane = gid & 31;
  int tk = (gid >> 5) % ktiles;
  int tn = (gid >> 5) / ktiles;
  int n  = tn*16 + (lane & 15);
  int kb = tk*32 + (lane >> 4)*16;
  int Kreal = Cin*5;
  __bf16* d = dst + (size_t)gid*16;
  for (int e = 0; e < 16; ++e){
    int k = kb + e;
    float v = 0.f;
    if (k < Kreal) v = w[((size_t)n*Cin + k/5)*5 + (k%5)];
    d[e] = (__bf16)v;
  }
}

// ---------------- per-step kernels ----------------

// blocks 0..31: location-sensitive attention (one batch row each, 128 threads)
// blocks 32..63: prenet
__global__ void k_pre(const float* __restrict__ hs, const float* __restrict__ enc,
                      const float* __restrict__ z0, const float* __restrict__ Wdec,
                      const float* __restrict__ Wcv, const float* __restrict__ Wloc,
                      const float* __restrict__ vv, float* __restrict__ aw,
                      float* __restrict__ attc,
                      const float* __restrict__ ys, const float* __restrict__ Wp0,
                      const float* __restrict__ bp0, const float* __restrict__ Wp1,
                      const float* __restrict__ bp1, float* __restrict__ pnb, int t){
  int tid = threadIdx.x;
  if (blockIdx.x < B_) {
    int b = blockIdx.x;
    __shared__ float aw_s[TENC], dec_s[ATT], red[TENC], aw2_s[TENC];
    aw_s[tid] = aw[b*TENC + tid];
    __syncthreads();
    // dec = z0[b] @ Wdec (thread owns column a = tid)
    float d = 0.f;
    for (int h = 0; h < HID; ++h) d += z0[b*HID + h] * Wdec[h*ATT + tid];
    dec_s[tid] = d;
    __syncthreads();
    // conv features for position t = tid
    float lf[CCH];
    for (int c = 0; c < CCH; ++c){
      float s = 0.f;
      for (int k = 0; k < KATT; ++k){
        int ts = tid + k - KATT/2;
        if (ts >= 0 && ts < TENC) s += aw_s[ts] * Wcv[c*KATT + k];
      }
      lf[c] = s;
    }
    float e = 0.f;
    for (int a = 0; a < ATT; ++a){
      float loc = 0.f;
      for (int c = 0; c < CCH; ++c) loc += lf[c] * Wloc[c*ATT + a];
      e += tanhf(enc[((size_t)b*TENC + tid)*ATT + a] + dec_s[a] + loc) * vv[a];
    }
    // sharpened softmax
    red[tid] = e; __syncthreads();
    for (int s = TENC/2; s > 0; s >>= 1){ if (tid < s) red[tid] = fmaxf(red[tid], red[tid+s]); __syncthreads(); }
    float mx = red[0]; __syncthreads();
    float w = __expf(SCAL*(e - mx));
    red[tid] = w; __syncthreads();
    for (int s = TENC/2; s > 0; s >>= 1){ if (tid < s) red[tid] += red[tid+s]; __syncthreads(); }
    float a2 = w / red[0];
    aw2_s[tid] = a2;
    aw[b*TENC + tid] = a2;
    __syncthreads();
    // context vector
    for (int d2 = tid; d2 < IDIM; d2 += TENC){
      float s = 0.f;
      for (int tt = 0; tt < TENC; ++tt) s += aw2_s[tt] * hs[((size_t)b*TENC + tt)*IDIM + d2];
      attc[b*IDIM + d2] = s;
    }
  } else {
    int b = blockIdx.x - B_;
    __shared__ float y_s[ODIM], h1_s[PRE];
    if (tid < ODIM) y_s[tid] = (t > 0) ? ys[((size_t)b*TOUT + (t-1))*ODIM + tid] : 0.f;
    __syncthreads();
    for (int p = tid; p < PRE; p += blockDim.x){
      float s = bp0[p];
      for (int o = 0; o < ODIM; ++o) s += y_s[o] * Wp0[o*PRE + p];
      h1_s[p] = fmaxf(s, 0.f);
    }
    __syncthreads();
    for (int p = tid; p < PRE; p += blockDim.x){
      float s = bp1[p];
      for (int q = 0; q < PRE; ++q) s += h1_s[q] * Wp1[q*PRE + p];
      pnb[b*PRE + p] = fmaxf(s, 0.f);
    }
  }
}

// Pack activation A fragments (M=32 rows = batch) from up to 3 concatenated fp32 sources.
// A k order chosen here (same as B packs): lane l -> row m = l&15; K(e) = (l>>4)*16 + e.
__global__ void k_packA_lstm(__bf16* dst, const float* S0, int l0, const float* S1, int l1,
                             const float* S2, int l2, int ktiles){
  int gid = blockIdx.x*blockDim.x + threadIdx.x;
  int total = 2*ktiles*32;
  if (gid >= total) return;
  int lane = gid & 31;
  int tk = (gid >> 5) % ktiles;
  int tm = (gid >> 5) / ktiles;
  int m  = tm*16 + (lane & 15);
  int b0 = (lane < 16) ? 0 : 8;
  __bf16* d = dst + (size_t)gid*16;
  for (int e = 0; e < 16; ++e){
    int kk = (lane >> 4)*16 + e;
    int k  = tk*32 + kk;
    float v;
    if      (k < l0)      v = S0[(size_t)m*l0 + k];
    else if (k < l0 + l1) v = S1[(size_t)m*l1 + (k - l0)];
    else                  v = S2[(size_t)m*l2 + (k - l0 - l1)];
    d[e] = (__bf16)v;
  }
}

// Fused LSTM layer: GEMM g = [x|h] @ [Wih|Whh]^T via WMMA bf16, then gates +
// zoneout in-block. Block covers 64 hidden units x all 4 gates x both M-tiles.
// 16 waves: wave w -> gate = w>>2, htile = w&3; grid = 16 blocks (H/64).
__global__ __launch_bounds__(512) void k_lstm(const __bf16* __restrict__ Apk,
    const __bf16* __restrict__ Wpk, int ktiles,
    const float* __restrict__ bc, float* __restrict__ z, float* __restrict__ c){
  __shared__ float gbuf[32][257];                 // 32 batch x 256 cols (+pad)
  int lane = threadIdx.x & 31;
  int wv   = threadIdx.x >> 5;
  int gate = wv >> 2, ht = wv & 3;
  int ntile = gate*64 + blockIdx.x*4 + ht;        // column tile in 4096-wide gate space
  const bf16x16* A = (const bf16x16*)Apk;
  const bf16x16* W = (const bf16x16*)Wpk;
  const bf16x16* a0p  = A + lane;
  const bf16x16* a1p  = A + (size_t)ktiles*32 + lane;
  const bf16x16* wrow = W + (size_t)ntile*ktiles*32 + lane;
  floatx8 acc0 = {0,0,0,0,0,0,0,0};
  floatx8 acc1 = {0,0,0,0,0,0,0,0};
  for (int kt = 0; kt < ktiles; ++kt){
    bf16x16 bf = wrow[kt*32];
    acc0 = wmma_bf16(a0p[kt*32], bf, acc0);
    acc1 = wmma_bf16(a1p[kt*32], bf, acc1);
  }
  // D layout: lanes 0-15 hold M=r, lanes 16-31 hold M=8+r; N = lane&15
  int nl = lane & 15, rb = (lane >> 4) * 8;
  int col = gate*64 + ht*16 + nl;
  #pragma unroll
  for (int r = 0; r < 8; ++r){
    gbuf[rb + r][col]      = acc0[r];
    gbuf[16 + rb + r][col] = acc1[r];
  }
  __syncthreads();
  for (int e = threadIdx.x; e < B_*64; e += 512){
    int m = e >> 6, j = e & 63;
    int h = blockIdx.x*64 + j;
    float gi = gbuf[m][j      ] + bc[h];
    float gf = gbuf[m][64  + j] + bc[HID   + h];
    float gn = gbuf[m][128 + j] + bc[2*HID + h];
    float go = gbuf[m][192 + j] + bc[3*HID + h];
    float cp = c[m*HID + h], zp = z[m*HID + h];
    float c2 = sigmf(gf)*cp + sigmf(gi)*tanhf(gn);
    float h2 = sigmf(go)*tanhf(c2);
    z[m*HID + h] = ZONE*zp + (1.f - ZONE)*h2;
    c[m*HID + h] = ZONE*cp + (1.f - ZONE)*c2;
  }
}

// feat_out: before[b, o, t] = [z1 | att_c] @ Wfeat
__global__ void k_out(const float* __restrict__ z1, const float* __restrict__ attc,
                      const float* __restrict__ Wf, float* __restrict__ before, int t){
  int b = blockIdx.x, o = threadIdx.x;
  float s = 0.f;
  for (int k = 0; k < HID;  ++k) s += z1[b*HID + k]   * Wf[k*ODIM + o];
  for (int k = 0; k < IDIM; ++k) s += attc[b*IDIM + k]* Wf[(HID + k)*ODIM + o];
  before[((size_t)b*ODIM + o)*TOUT + t] = s;
}

// ---------------- postnet ----------------

// im2col A-fragment pack: M = b*256 + tau (8192 rows), K = Cin*5 (zero padded)
__global__ void k_packA_post(__bf16* dst, const float* __restrict__ x, int Cin, int ktiles){
  int gid = blockIdx.x*blockDim.x + threadIdx.x;
  int total = 512*ktiles*32;
  if (gid >= total) return;
  int lane = gid & 31;
  int tk = (gid >> 5) % ktiles;
  int tm = (gid >> 5) / ktiles;
  int m  = tm*16 + (lane & 15);
  int b  = m >> 8, tau = m & 255;
  int b0 = (lane < 16) ? 0 : 8;
  int Kreal = Cin*5;
  __bf16* d = dst + (size_t)gid*16;
  for (int e = 0; e < 16; ++e){
    int kk = (lane >> 4)*16 + e;
    int k  = tk*32 + kk;
    float v = 0.f;
    if (k < Kreal){
      int ci = k/5, q = k%5, ts = tau + q - 2;
      if (ts >= 0 && ts < TOUT) v = x[((size_t)b*Cin + ci)*TOUT + ts];
    }
    d[e] = (__bf16)v;
  }
}

// generic WMMA GEMM: out(8192 x N) = Apk(8192 x K) * Wpk(K x N); one tile per wave
__global__ __launch_bounds__(256) void k_gemm_post(const __bf16* __restrict__ Apk,
    const __bf16* __restrict__ Wpk, float* __restrict__ out, int ntilesN, int ktiles){
  int lane = threadIdx.x & 31;
  int wg = blockIdx.x*8 + (threadIdx.x >> 5);
  int mtile = wg / ntilesN, ntile = wg % ntilesN;
  const bf16x16* A = (const bf16x16*)Apk + (size_t)mtile*ktiles*32 + lane;
  const bf16x16* W = (const bf16x16*)Wpk + (size_t)ntile*ktiles*32 + lane;
  floatx8 acc = {0,0,0,0,0,0,0,0};
  for (int kt = 0; kt < ktiles; ++kt)
    acc = wmma_bf16(A[kt*32], W[kt*32], acc);
  int N = ntilesN*16;
  int nl = lane & 15, rb = (lane >> 4)*8;
  #pragma unroll
  for (int r = 0; r < 8; ++r)
    out[(size_t)(mtile*16 + rb + r)*N + ntile*16 + nl] = acc[r];
}

// BN(eval) + tanh (+ residual) epilogue; dst in (B, Cout, T) layout
__global__ void k_post_epi(const float* __restrict__ gm, const float* __restrict__ gamma,
    const float* __restrict__ beta, const float* __restrict__ resid,
    float* __restrict__ dst, int Cout, int do_tanh){
  int gid = blockIdx.x*blockDim.x + threadIdx.x;
  int total = B_*Cout*TOUT;
  if (gid >= total) return;
  int tau = gid & 255;
  int rem = gid >> 8;
  int co = rem % Cout, b = rem / Cout;
  float v = gm[((size_t)(b*TOUT + tau))*Cout + co] * (gamma[co] * rsqrtf(1.0f + 1e-5f)) + beta[co];
  if (do_tanh) v = tanhf(v);
  if (resid) v += resid[gid];
  dst[gid] = v;
}

// ---------------- host driver ----------------

extern "C" void kernel_launch(void* const* d_in, const int* in_sizes, int n_in,
                              void* d_out, int out_size, void* d_ws, size_t ws_size,
                              hipStream_t stream){
  (void)in_sizes; (void)n_in; (void)out_size; (void)ws_size;
  const float* hs   = (const float*)d_in[0];
  const float* ys   = (const float*)d_in[1];
  const float* Wp0  = (const float*)d_in[2];
  const float* bp0  = (const float*)d_in[3];
  const float* Wp1  = (const float*)d_in[4];
  const float* bp1  = (const float*)d_in[5];
  const float* Wenc = (const float*)d_in[6];
  const float* benc = (const float*)d_in[7];
  const float* Wdec = (const float*)d_in[8];
  const float* Wcv  = (const float*)d_in[9];
  const float* Wloc = (const float*)d_in[10];
  const float* vv   = (const float*)d_in[11];
  const float* Wih0 = (const float*)d_in[12];
  const float* Whh0 = (const float*)d_in[13];
  const float* bih0 = (const float*)d_in[14];
  const float* bhh0 = (const float*)d_in[15];
  const float* Wih1 = (const float*)d_in[16];
  const float* Whh1 = (const float*)d_in[17];
  const float* bih1 = (const float*)d_in[18];
  const float* bhh1 = (const float*)d_in[19];
  const float* Wfeat= (const float*)d_in[20];
  const float* pc[5]; for (int l = 0; l < 5; ++l) pc[l] = (const float*)d_in[21+l];
  const float* gg[5]; for (int l = 0; l < 5; ++l) gg[l] = (const float*)d_in[26+l];
  const float* bb[5]; for (int l = 0; l < 5; ++l) bb[l] = (const float*)d_in[31+l];

  char* ws = (char*)d_ws; size_t off = 0;
  auto alloc = [&](size_t bytes)->void*{
    void* p = ws + off; off = (off + bytes + 255) & ~(size_t)255; return p;
  };

  const int KT0 = 56;   // (512+256+1024)/32
  const int KT1 = 64;   // (1024+1024)/32
  const int post_nt[5]  = {32, 32, 32, 32, 5};
  const int post_kt[5]  = {13, 80, 80, 80, 80};
  const int post_cin[5] = {80, 512, 512, 512, 512};

  __bf16* WPK0 = (__bf16*)alloc((size_t)256*KT0*512*2);
  __bf16* WPK1 = (__bf16*)alloc((size_t)256*KT1*512*2);
  __bf16* WPP[5];
  for (int l = 0; l < 5; ++l) WPP[l] = (__bf16*)alloc((size_t)post_nt[l]*post_kt[l]*512*2);
  float* ENC = (float*)alloc((size_t)B_*TENC*ATT*4);
  float* Z0  = (float*)alloc((size_t)B_*HID*4);
  float* C0  = (float*)alloc((size_t)B_*HID*4);
  float* Z1  = (float*)alloc((size_t)B_*HID*4);
  float* C1  = (float*)alloc((size_t)B_*HID*4);
  float* AW  = (float*)alloc((size_t)B_*TENC*4);
  float* ATC = (float*)alloc((size_t)B_*IDIM*4);
  float* PNB = (float*)alloc((size_t)B_*PRE*4);
  float* BC0 = (float*)alloc((size_t)4*HID*4);
  float* BC1 = (float*)alloc((size_t)4*HID*4);
  __bf16* APK0 = (__bf16*)alloc((size_t)2*KT0*512*2);
  __bf16* APK1 = (__bf16*)alloc((size_t)2*KT1*512*2);
  __bf16* APST = (__bf16*)alloc((size_t)512*80*512*2);
  float* GOUT  = (float*)alloc((size_t)8192*512*4);
  float* XB0   = (float*)alloc((size_t)B_*512*TOUT*4);
  float* XB1   = (float*)alloc((size_t)B_*512*TOUT*4);

  float* AFTER  = (float*)d_out;
  float* BEFORE = (float*)d_out + (size_t)B_*ODIM*TOUT;

  // one-time init + weight packing (re-done each call; deterministic)
  k_init<<<128, 256, 0, stream>>>(Z0, C0, Z1, C1, AW);
  k_biascomb<<<16, 256, 0, stream>>>(bih0, bhh0, bih1, bhh1, BC0, BC1);
  k_encproj<<<B_*TENC, ATT, 0, stream>>>(hs, Wenc, benc, ENC);
  { int tot = 256*KT0*32; k_packW_lstm<<<(tot+255)/256, 256, 0, stream>>>(WPK0, Wih0, 768,  Whh0, 1024, 256, KT0); }
  { int tot = 256*KT1*32; k_packW_lstm<<<(tot+255)/256, 256, 0, stream>>>(WPK1, Wih1, 1024, Whh1, 1024, 256, KT1); }
  for (int l = 0; l < 5; ++l){
    int tot = post_nt[l]*post_kt[l]*32;
    k_packW_conv<<<(tot+255)/256, 256, 0, stream>>>(WPP[l], pc[l], post_cin[l], post_nt[l], post_kt[l]);
  }

  // sequential decoder scan
  for (int t = 0; t < TOUT; ++t){
    k_pre<<<2*B_, TENC, 0, stream>>>(hs, ENC, Z0, Wdec, Wcv, Wloc, vv, AW, ATC,
                                     ys, Wp0, bp0, Wp1, bp1, PNB, t);
    { int tot = 2*KT0*32; k_packA_lstm<<<(tot+255)/256, 256, 0, stream>>>(APK0, ATC, IDIM, PNB, PRE, Z0, HID, KT0); }
    k_lstm<<<16, 512, 0, stream>>>(APK0, WPK0, KT0, BC0, Z0, C0);
    { int tot = 2*KT1*32; k_packA_lstm<<<(tot+255)/256, 256, 0, stream>>>(APK1, Z0, HID, Z1, HID, (const float*)nullptr, 0, KT1); }
    k_lstm<<<16, 512, 0, stream>>>(APK1, WPK1, KT1, BC1, Z1, C1);
    k_out<<<B_, ODIM, 0, stream>>>(Z1, ATC, Wfeat, BEFORE, t);
  }

  // postnet: im2col pack -> WMMA GEMM -> BN/tanh (+residual on last layer)
  const float* xin = BEFORE;
  float* xb[2] = {XB0, XB1};
  for (int l = 0; l < 5; ++l){
    int kt = post_kt[l], nt = post_nt[l], cout = nt*16;
    { int tot = 512*kt*32; k_packA_post<<<(tot+255)/256, 256, 0, stream>>>(APST, xin, post_cin[l], kt); }
    { int tiles = 512*nt;  k_gemm_post<<<tiles/8, 256, 0, stream>>>(APST, WPP[l], GOUT, nt, kt); }
    float* dst = (l < 4) ? xb[l & 1] : AFTER;
    const float* res = (l == 4) ? BEFORE : nullptr;
    { int tot = B_*cout*TOUT; k_post_epi<<<(tot+255)/256, 256, 0, stream>>>(GOUT, gg[l], bb[l], res, dst, cout, (l < 4) ? 1 : 0); }
    xin = dst;
  }
}
